// CognateRefiner_89489938580158
// MI455X (gfx1250) — matrix operand order repaired and checked by reading, not compiled, not run on hardware
//
#include <hip/hip_runtime.h>
#include <hip/hip_bf16.h>
#include <math.h>

typedef __attribute__((ext_vector_type(16))) _Float16 v16h;
typedef __attribute__((ext_vector_type(8)))  float    v8f;

typedef unsigned int       u32;
typedef unsigned long long u64;
typedef u32 __attribute__((ext_vector_type(4))) u32x4;
typedef int __attribute__((ext_vector_type(4))) i32x4;
typedef int __attribute__((ext_vector_type(8))) i32x8;

constexpr int V  = 32000, D = 216, NL = 11, NH = 4, HD = 54, FF = 864;
constexpr int Bb = 2, S = 1024, BS = Bb * S;
constexpr float EPS = 1e-5f;
constexpr float THETA = 10000.0f;

// padded sizes (multiples of 64 so GEMM tiles need no bounds checks)
constexpr int KpD  = 256;   // K pad for K=216
constexpr int KpFF = 896;   // K pad for K=864
constexpr int Np_QKV = 704; // N pad for 648
constexpr int Np_D   = 256; // N pad for 216
constexpr int Np_FF  = 896; // N pad for 864

#if defined(__has_builtin)
#if __has_builtin(__builtin_amdgcn_tensor_load_to_lds) && __has_builtin(__builtin_amdgcn_s_wait_tensorcnt)
#define HAVE_TDM 1
#else
#define HAVE_TDM 0
#endif
#else
#define HAVE_TDM 0
#endif

// ---------------------------------------------------------------------------
// TDM: issue an async 2D tile load (rows x cols f16, row stride in elements)
// from global memory into LDS. D# layout per CDNA5 ISA §8.3/8.4:
//   group0: count=1 | lds_addr | global_addr[56:0] | type=2
//   group1: data_size=1(2B), tensor_dim0=cols, tensor_dim1=rows,
//           tile_dim0=cols, tile_dim1=rows, tensor_dim0_stride=stride
//   groups 2/3 (+spare group): zero (2D tensor)
// This toolchain exposes the 6-arg builtin:
//   (uint32x4 g0, int32x8 g1, int32x4, int32x4, int32x8, i32 cpol)
// ---------------------------------------------------------------------------
#if HAVE_TDM
__device__ __forceinline__ void tdm_load_tile_f16(u32 lds_byte,
                                                  const _Float16* gptr,
                                                  u32 rows, u32 cols,
                                                  u32 stride_elems) {
  const u64 ga = (u64)(uintptr_t)gptr;
  u32x4 g0;
  g0.x = 1u;                                              // count=1, user desc
  g0.y = lds_byte;                                        // LDS dest (bytes)
  g0.z = (u32)ga;                                         // global addr lo
  g0.w = (u32)((ga >> 32) & 0x01FFFFFFu) | (2u << 30);    // addr[56:32], type=2
  i32x8 g1;
  g1[0] = (int)(1u << 16);                                // data_size=1 (2B)
  g1[1] = (int)((cols & 0xFFFFu) << 16);                  // tensor_dim0 lo16
  g1[2] = (int)(((cols >> 16) & 0xFFFFu) |                // tensor_dim0 hi16
                ((rows & 0xFFFFu) << 16));                // tensor_dim1 lo16
  g1[3] = (int)(((rows >> 16) & 0xFFFFu) |                // tensor_dim1 hi16
                ((cols & 0xFFFFu) << 16));                // tile_dim0
  g1[4] = (int)(rows & 0xFFFFu);                          // tile_dim1, tile_dim2=0
  g1[5] = (int)stride_elems;                              // tensor_dim0_stride lo32
  g1[6] = 0;
  g1[7] = 0;
  const i32x4 z4 = {0, 0, 0, 0};
  const i32x8 z8 = {0, 0, 0, 0, 0, 0, 0, 0};
  __builtin_amdgcn_tensor_load_to_lds(g0, g1, z4, z4, z8, 0);
}
#endif

// ---------------------------------------------------------------------------
// Embedding gather
// ---------------------------------------------------------------------------
__global__ void embed_kernel(const int* __restrict__ ids,
                             const float* __restrict__ embed,
                             float* __restrict__ x) {
  const int row = blockIdx.x;
  const int tok = ids[row];
  const float* src = embed + (size_t)tok * D;
  float* dst = x + (size_t)row * D;
  for (int d = threadIdx.x; d < D; d += blockDim.x) dst[d] = src[d];
}

// ---------------------------------------------------------------------------
// RMSNorm (one block per row)
// ---------------------------------------------------------------------------
__global__ __launch_bounds__(256) void rmsnorm_kernel(const float* __restrict__ x,
                                                      const float* __restrict__ w,
                                                      float* __restrict__ out, int dim) {
  const int row = blockIdx.x;
  const float* xr = x + (size_t)row * dim;
  float* orow = out + (size_t)row * dim;
  __shared__ float red[256];
  float s = 0.f;
  for (int d = threadIdx.x; d < dim; d += blockDim.x) { float v = xr[d]; s += v * v; }
  red[threadIdx.x] = s;
  __syncthreads();
  for (int off = 128; off > 0; off >>= 1) {
    if (threadIdx.x < off) red[threadIdx.x] += red[threadIdx.x + off];
    __syncthreads();
  }
  const float inv = rsqrtf(red[0] / (float)dim + EPS);
  for (int d = threadIdx.x; d < dim; d += blockDim.x) orow[d] = xr[d] * inv * w[d];
}

// ---------------------------------------------------------------------------
// fp32 -> f16 with zero padding: dst is (gridDim.x rows) x Kp, src is rows x K.
// Rows >= rows and cols >= K are zero-filled, so GEMM tiles are guard-free.
// ---------------------------------------------------------------------------
__global__ __launch_bounds__(256) void f16_pad_kernel(const float* __restrict__ src,
                                                      _Float16* __restrict__ dst,
                                                      int rows, int K, int Kp) {
  const int r = blockIdx.x;
  _Float16* drow = dst + (size_t)r * Kp;
  if (r < rows) {
    const float* srow = src + (size_t)r * K;
    for (int k = threadIdx.x; k < Kp; k += blockDim.x)
      drow[k] = (k < K) ? (_Float16)srow[k] : (_Float16)0.f;
  } else {
    for (int k = threadIdx.x; k < Kp; k += blockDim.x) drow[k] = (_Float16)0.f;
  }
}

// ---------------------------------------------------------------------------
// WMMA GEMM:  C[M,N] = A[M,Kp] * W[Np,Kp]^T  (+ optional residual R[M,N])
// A, W are pre-converted zero-padded f16. 64x64 tile, 128 threads = 4 waves,
// double-buffered LDS with TDM (tensor_load_to_lds) prefetch of the next
// K-tile overlapped with WMMA compute; s_wait_tensorcnt + barrier to consume.
// ---------------------------------------------------------------------------
__global__ __launch_bounds__(128) void wmma_gemm_kernel(
    const _Float16* __restrict__ A,   // M x Kp
    const _Float16* __restrict__ Wh,  // Np x Kp
    const float* __restrict__ R,      // optional residual (nullptr = none)
    float* __restrict__ C,            // M x N
    int M, int N, int Kp) {
  __shared__ __align__(16) _Float16 sA[2][64 * 64];
  __shared__ __align__(16) _Float16 sB[2][64 * 64];

  const int tid  = threadIdx.x;
  const int lane = tid & 31;
  const int wave = tid >> 5;
  const int hi   = (lane >= 16) ? 1 : 0;
  const int ln   = lane & 15;
  const int tileM = blockIdx.y * 64;
  const int tileN = blockIdx.x * 64;

  const _Float16* Abase = A + (size_t)tileM * Kp;
  const _Float16* Wbase = Wh + (size_t)tileN * Kp;

  v8f acc[4] = {};

  // ---- stage first K-tile into buffer 0 ----
#if HAVE_TDM
  if (tid < 32) {
    tdm_load_tile_f16((u32)(uintptr_t)&sA[0][0], Abase, 64, 64, (u32)Kp);
    tdm_load_tile_f16((u32)(uintptr_t)&sB[0][0], Wbase, 64, 64, (u32)Kp);
    __builtin_amdgcn_s_wait_tensorcnt((short)0);
  }
#else
  #pragma unroll
  for (int p = 0; p < 4; ++p) {
    const int idx = tid + (p << 7);
    const int row = idx >> 3, c8 = idx & 7;
    *(uint4*)(&sA[0][row * 64 + c8 * 8]) =
        *(const uint4*)(Abase + (size_t)row * Kp + c8 * 8);
    *(uint4*)(&sB[0][row * 64 + c8 * 8]) =
        *(const uint4*)(Wbase + (size_t)row * Kp + c8 * 8);
  }
#endif
  __syncthreads();

  int buf = 0;
  for (int kk = 0; kk < Kp; kk += 64) {
    const int nxt = buf ^ 1;
    const bool more = (kk + 64 < Kp);

    // ---- prefetch next K-tile into the other buffer (overlaps WMMA) ----
    if (more) {
#if HAVE_TDM
      if (tid < 32) {
        tdm_load_tile_f16((u32)(uintptr_t)&sA[nxt][0], Abase + kk + 64, 64, 64, (u32)Kp);
        tdm_load_tile_f16((u32)(uintptr_t)&sB[nxt][0], Wbase + kk + 64, 64, 64, (u32)Kp);
      }
#else
      #pragma unroll
      for (int p = 0; p < 4; ++p) {
        const int idx = tid + (p << 7);
        const int row = idx >> 3, c8 = idx & 7;
        *(uint4*)(&sA[nxt][row * 64 + c8 * 8]) =
            *(const uint4*)(Abase + (size_t)row * Kp + kk + 64 + c8 * 8);
        *(uint4*)(&sB[nxt][row * 64 + c8 * 8]) =
            *(const uint4*)(Wbase + (size_t)row * Kp + kk + 64 + c8 * 8);
      }
#endif
    }

    // ---- compute on current buffer: 2 k-steps of 32, 4 col-blocks ----
    const _Float16* tA = &sA[buf][0];
    const _Float16* tB = &sB[buf][0];
    const int arow = (wave << 4) + ln;
    #pragma unroll
    for (int t = 0; t < 2; ++t) {
      const int kb = t << 5;
      // A fragment (16x32): lanes 0-15 K{0..7,16..23}, lanes 16-31 K{8..15,24..31}
      v16h af;
      #pragma unroll
      for (int i = 0; i < 16; ++i) {
        const int k = kb + ((i < 8) ? ((hi ? 8 : 0) + i)
                                    : (16 + (hi ? 8 : 0) + (i - 8)));
        af[i] = tA[arow * 64 + k];
      }
      #pragma unroll
      for (int cblk = 0; cblk < 4; ++cblk) {
        // B fragment (32x16): lanes 0-15 K{0..7,16..23}, lanes 16-31 K{8..15,24..31}
        v16h bf;
        const int bcol = (cblk << 4) + ln;
        #pragma unroll
        for (int i = 0; i < 16; ++i) {
          const int k = kb + ((i < 8) ? ((hi ? 8 : 0) + i) : (16 + (hi ? 8 : 0) + (i - 8)));
          bf[i] = tB[bcol * 64 + k];
        }
        acc[cblk] = __builtin_amdgcn_wmma_f32_16x16x32_f16(
            false, af, false, bf, (short)0, acc[cblk], false, false);
      }
    }

#if HAVE_TDM
    if (more && tid < 32) __builtin_amdgcn_s_wait_tensorcnt((short)0);
#endif
    __syncthreads();
    buf = nxt;
  }

  // ---- store: 16x16 f32 C/D layout: VGPR r -> M = r + 8*hi, N = lane%16 ----
  #pragma unroll
  for (int cblk = 0; cblk < 4; ++cblk) {
    #pragma unroll
    for (int r = 0; r < 8; ++r) {
      const int gm = tileM + (wave << 4) + r + (hi ? 8 : 0);
      const int gn = tileN + (cblk << 4) + ln;
      if (gm < M && gn < N) {
        float v = acc[cblk][r];
        if (R) v += R[(size_t)gm * N + gn];
        C[(size_t)gm * N + gn] = v;
      }
    }
  }
}

// ---------------------------------------------------------------------------
// RoPE in place on q,k halves of qkv (one block per row)
// ---------------------------------------------------------------------------
__global__ void rope_kernel(float* __restrict__ qkv) {
  const int row = blockIdx.x;
  const int s = row % S;
  float* base = qkv + (size_t)row * (3 * D);
  constexpr int HALF = HD / 2;
  for (int t = threadIdx.x; t < 2 * NH * HALF; t += blockDim.x) {
    const int which = t / (NH * HALF);
    const int rem = t % (NH * HALF);
    const int h = rem / HALF;
    const int j = rem % HALF;
    const float invf = powf(THETA, -2.0f * (float)j / (float)HD);
    const float ang = (float)s * invf;
    const float c = cosf(ang), sn = sinf(ang);
    float* p = base + which * D + h * HD;
    const float x1 = p[j], x2 = p[j + HALF];
    p[j]        = x1 * c - x2 * sn;
    p[j + HALF] = x2 * c + x1 * sn;
  }
}

// ---------------------------------------------------------------------------
// Causal attention, one block per (b, h, q)
// ---------------------------------------------------------------------------
__global__ __launch_bounds__(64) void attn_kernel(const float* __restrict__ qkv,
                                                  float* __restrict__ out) {
  const int q = blockIdx.x, h = blockIdx.y, b = blockIdx.z;
  const int tid = threadIdx.x;
  __shared__ float sc[S];
  __shared__ float red[64];
  const float* qrow = qkv + ((size_t)(b * S + q) * (3 * D)) + h * HD;
  const float scale = rsqrtf((float)HD);

  for (int k = tid; k <= q; k += 64) {
    const float* krow = qkv + ((size_t)(b * S + k) * (3 * D)) + D + h * HD;
    float dot = 0.f;
    #pragma unroll
    for (int d = 0; d < HD; ++d) dot += qrow[d] * krow[d];
    sc[k] = dot * scale;
  }
  __syncthreads();

  float m = -3.0e38f;
  for (int k = tid; k <= q; k += 64) m = fmaxf(m, sc[k]);
  red[tid] = m; __syncthreads();
  for (int off = 32; off > 0; off >>= 1) {
    if (tid < off) red[tid] = fmaxf(red[tid], red[tid + off]);
    __syncthreads();
  }
  m = red[0]; __syncthreads();

  float ssum = 0.f;
  for (int k = tid; k <= q; k += 64) { float e = __expf(sc[k] - m); sc[k] = e; ssum += e; }
  red[tid] = ssum; __syncthreads();
  for (int off = 32; off > 0; off >>= 1) {
    if (tid < off) red[tid] += red[tid + off];
    __syncthreads();
  }
  const float inv = 1.0f / red[0];
  __syncthreads();

  for (int d = tid; d < HD; d += 64) {
    float acc = 0.f;
    for (int k = 0; k <= q; ++k)
      acc += sc[k] * qkv[((size_t)(b * S + k) * (3 * D)) + 2 * D + h * HD + d];
    out[(size_t)(b * S + q) * D + h * HD + d] = acc * inv;
  }
}

// ---------------------------------------------------------------------------
// gate = silu(gate) * up
// ---------------------------------------------------------------------------
__global__ void silu_mul_kernel(float* __restrict__ gate,
                                const float* __restrict__ up, int n) {
  const int i = blockIdx.x * blockDim.x + threadIdx.x;
  if (i < n) {
    const float g = gate[i];
    gate[i] = g * (1.0f / (1.0f + __expf(-g))) * up[i];
  }
}

// ---------------------------------------------------------------------------
extern "C" void kernel_launch(void* const* d_in, const int* in_sizes, int n_in,
                              void* d_out, int out_size, void* d_ws, size_t ws_size,
                              hipStream_t stream) {
  const int*   ids    = (const int*)  d_in[0];
  const float* embed  = (const float*)d_in[1];
  const float* ln1_w  = (const float*)d_in[2];
  const float* qkv_w  = (const float*)d_in[3];
  const float* o_w    = (const float*)d_in[4];
  const float* ln2_w  = (const float*)d_in[5];
  const float* gate_w = (const float*)d_in[6];
  const float* up_w   = (const float*)d_in[7];
  const float* down_w = (const float*)d_in[8];
  const float* norm_w = (const float*)d_in[9];
  const float* lm_w   = (const float*)d_in[10];
  float* logits = (float*)d_out;

  // fp32 workspace
  float* x    = (float*)d_ws;                       // BS*D
  float* hbuf = x    + (size_t)BS * D;              // BS*D
  float* qkv  = hbuf + (size_t)BS * D;              // BS*3D
  float* att  = qkv  + (size_t)BS * 3 * D;          // BS*D
  float* gate = att  + (size_t)BS * D;              // BS*FF
  float* up   = gate + (size_t)BS * FF;             // BS*FF
  // f16 padded staging buffers
  _Float16* aF16 = (_Float16*)(up + (size_t)BS * FF);   // BS x KpFF
  _Float16* wF16 = aF16 + (size_t)BS * KpFF;            // V x KpD (largest W)

  auto cdiv = [](int a, int b) { return (a + b - 1) / b; };
  const dim3 gemmBlock(128);

  embed_kernel<<<BS, 256, 0, stream>>>(ids, embed, x);

  for (int l = 0; l < NL; ++l) {
    // --- attention block ---
    rmsnorm_kernel<<<BS, 256, 0, stream>>>(x, ln1_w + (size_t)l * D, hbuf, D);
    f16_pad_kernel<<<BS, 256, 0, stream>>>(hbuf, aF16, BS, D, KpD);
    f16_pad_kernel<<<Np_QKV, 256, 0, stream>>>(qkv_w + (size_t)l * 3 * D * D, wF16,
                                               3 * D, D, KpD);
    wmma_gemm_kernel<<<dim3(cdiv(3 * D, 64), BS / 64), gemmBlock, 0, stream>>>(
        aF16, wF16, nullptr, qkv, BS, 3 * D, KpD);

    rope_kernel<<<BS, 224, 0, stream>>>(qkv);
    attn_kernel<<<dim3(S, NH, Bb), 64, 0, stream>>>(qkv, att);

    f16_pad_kernel<<<BS, 256, 0, stream>>>(att, aF16, BS, D, KpD);
    f16_pad_kernel<<<Np_D, 256, 0, stream>>>(o_w + (size_t)l * D * D, wF16, D, D, KpD);
    wmma_gemm_kernel<<<dim3(cdiv(D, 64), BS / 64), gemmBlock, 0, stream>>>(
        aF16, wF16, x, x, BS, D, KpD);   // x += att @ o_w^T

    // --- MLP block ---
    rmsnorm_kernel<<<BS, 256, 0, stream>>>(x, ln2_w + (size_t)l * D, hbuf, D);
    f16_pad_kernel<<<BS, 256, 0, stream>>>(hbuf, aF16, BS, D, KpD);

    f16_pad_kernel<<<Np_FF, 256, 0, stream>>>(gate_w + (size_t)l * FF * D, wF16,
                                              FF, D, KpD);
    wmma_gemm_kernel<<<dim3(cdiv(FF, 64), BS / 64), gemmBlock, 0, stream>>>(
        aF16, wF16, nullptr, gate, BS, FF, KpD);

    f16_pad_kernel<<<Np_FF, 256, 0, stream>>>(up_w + (size_t)l * FF * D, wF16,
                                              FF, D, KpD);
    wmma_gemm_kernel<<<dim3(cdiv(FF, 64), BS / 64), gemmBlock, 0, stream>>>(
        aF16, wF16, nullptr, up, BS, FF, KpD);

    silu_mul_kernel<<<cdiv(BS * FF, 256), 256, 0, stream>>>(gate, up, BS * FF);

    f16_pad_kernel<<<BS, 256, 0, stream>>>(gate, aF16, BS, FF, KpFF);
    f16_pad_kernel<<<Np_D, 256, 0, stream>>>(down_w + (size_t)l * D * FF, wF16,
                                             D, FF, KpFF);
    wmma_gemm_kernel<<<dim3(cdiv(D, 64), BS / 64), gemmBlock, 0, stream>>>(
        aF16, wF16, x, x, BS, D, KpFF);  // x += act @ down_w^T
  }

  // --- final norm + lm_head ---
  rmsnorm_kernel<<<BS, 256, 0, stream>>>(x, norm_w, hbuf, D);
  f16_pad_kernel<<<BS, 256, 0, stream>>>(hbuf, aF16, BS, D, KpD);
  f16_pad_kernel<<<V, 256, 0, stream>>>(lm_w, wF16, V, D, KpD);
  wmma_gemm_kernel<<<dim3(V / 64, BS / 64), gemmBlock, 0, stream>>>(
      aF16, wF16, nullptr, logits, BS, V, KpD);
}
